// Attention_23476291240718
// MI455X (gfx1250) — matrix operand order repaired and checked by reading, not compiled, not run on hardware
//
#include <hip/hip_runtime.h>

// ---------------------------------------------------------------------------
// Fused multi-head attention for MI455X (gfx1250, wave32, WMMA bf16 + TDM).
//   B=2, S=2048, H=16, D=64, D_IN=OUT_DIM=1024
// Phase 1: Qp = q@Wq, Kp = k@Wk  -> bf16 [B,H,S,D]
//          Vt = v@Wv             -> bf16 [B,H,D,S]   (transposed for P*V)
// Phase 2: flash attention per (b,h): K/V chunks DMA'd to LDS by the Tensor
//          Data Mover (double-buffered, TENSORcnt), QK^T and P*V on the WMMA
//          pipe, online softmax on half-wave shuffles, fp32 out.
// ---------------------------------------------------------------------------

typedef __bf16 bf16;
typedef bf16  v16bf __attribute__((ext_vector_type(16)));
typedef bf16  v8bf  __attribute__((ext_vector_type(8)));
typedef bf16  v4bf  __attribute__((ext_vector_type(4)));
typedef float v8f   __attribute__((ext_vector_type(8)));
typedef float v4f   __attribute__((ext_vector_type(4)));
typedef unsigned int u32x4 __attribute__((ext_vector_type(4)));
typedef int   i32x4 __attribute__((ext_vector_type(4)));
typedef int   i32x8 __attribute__((ext_vector_type(8)));

constexpr int BATCH = 2;
constexpr int SEQ   = 2048;
constexpr int NHEAD = 16;
constexpr int HDIM  = 64;
constexpr int DIN   = 1024;
constexpr int DOUT  = 1024;
constexpr int MROWS = BATCH * SEQ;   // 4096

#if defined(__has_builtin)
#if __has_builtin(__builtin_amdgcn_tensor_load_to_lds) && \
    __has_builtin(__builtin_amdgcn_s_wait_tensorcnt)
#define USE_TDM 1
#endif
#endif
#ifndef USE_TDM
#define USE_TDM 0
#endif

__device__ __forceinline__ bf16 f2bf(float x) {
    unsigned u = __builtin_bit_cast(unsigned, x);
    unsigned r = (u + 0x7FFFu + ((u >> 16) & 1u)) >> 16;   // RNE
    unsigned short s = (unsigned short)r;
    return __builtin_bit_cast(bf16, s);
}

__device__ __forceinline__ v16bf join8(v8bf lo, v8bf hi) {
    return __builtin_shufflevector(lo, hi, 0,1,2,3,4,5,6,7,8,9,10,11,12,13,14,15);
}

__device__ __forceinline__ v8f wmma_bf16(v16bf a, v16bf b, v8f c) {
    // (neg_a, A, neg_b, B, c_mod, C, reuse_a, reuse_b)
    return __builtin_amdgcn_wmma_f32_16x16x32_bf16(false, a, false, b,
                                                   (short)0, c, false, false);
}

#if USE_TDM
// --- Tensor DMA descriptor builders (CDNA5 ISA ch.8, D# groups 0/1) --------
__device__ __forceinline__ u32x4 tdm_group0(unsigned lds_byte_off,
                                            unsigned long long gaddr) {
    u32x4 g;
    g[0] = 1u;                                           // count=1 valid user D#
    g[1] = lds_byte_off;                                 // lds_addr (bytes)
    g[2] = (unsigned)(gaddr & 0xFFFFFFFFull);            // global_addr[31:0]
    g[3] = (unsigned)((gaddr >> 32) & 0x1FFFFFFull)      // global_addr[56:32]
         | (2u << 30);                                   // type = 2 ("image")
    return g;
}

// 2D tile of 16-bit elements: tile_dim0 x tile_dim1, row stride t0stride
__device__ __forceinline__ i32x8 tdm_group1_2d(unsigned tile0, unsigned tile1,
                                               unsigned long long t0stride,
                                               unsigned td0, unsigned td1) {
    i32x8 g;
    g[0] = (int)(1u << 16);                              // data_size=1 (2 bytes)
    g[1] = (int)((td0 & 0xFFFFu) << 16);                 // tensor_dim0[15:0]
    g[2] = (int)(((td0 >> 16) & 0xFFFFu) | ((td1 & 0xFFFFu) << 16));
    g[3] = (int)(((td1 >> 16) & 0xFFFFu) | ((tile0 & 0xFFFFu) << 16));
    g[4] = (int)(tile1 & 0xFFFFu);                       // tile_dim1, tile_dim2=0
    g[5] = (int)(t0stride & 0xFFFFFFFFull);              // dim0_stride[31:0]
    g[6] = (int)((t0stride >> 32) & 0xFFFFull);          // dim0_stride[47:32]
    g[7] = 0;
    return g;
}

__device__ __forceinline__ unsigned lds_offset_of(const void* p) {
    return (unsigned)(reinterpret_cast<uintptr_t>(p));   // low 32 bits = LDS off
}

__device__ __forceinline__ void tdm_load_2d(unsigned lds_byte_off,
                                            unsigned long long gaddr,
                                            i32x8 g1) {
    const i32x4 z4 = i32x4{0, 0, 0, 0};
    const i32x8 z8 = i32x8{0, 0, 0, 0, 0, 0, 0, 0};
    // 6-arg form (clang-23 / therock-10.0): groups 0..3 + extra group + cpol
    __builtin_amdgcn_tensor_load_to_lds(tdm_group0(lds_byte_off, gaddr),
                                        g1, z4, z4, z8, 0);
}
#endif

// ---------------------------------------------------------------------------
// Projection GEMM: Y = X[4096,1024] * W[1024,1024] -> bf16, scattered to
// [B,H,S,D] (TRANS=0) or [B,H,D,S] (TRANS=1, vectorized 16B stores).
// ---------------------------------------------------------------------------
template <int TRANS>
__global__ __launch_bounds__(256)
void proj_gemm(const float* __restrict__ X, const float* __restrict__ W,
               bf16* __restrict__ Y)
{
    __shared__ __align__(16) bf16 Alds[128 * 32];   // [row][k]
    __shared__ __align__(16) bf16 Blds[128 * 32];   // [col][k]  (W transposed)

    const int t    = threadIdx.x;
    const int lane = t & 31;
    const int wid  = t >> 5;
    const int lrow = lane & 15;
    const int g    = lane >> 4;           // half-wave group
    const int wm   = wid >> 1;            // 0..3
    const int wn   = wid & 1;             // 0..1
    const int m0   = blockIdx.y * 128;
    const int n0   = blockIdx.x * 128;

    v8f acc[2][4];
    #pragma unroll
    for (int mt = 0; mt < 2; ++mt)
        #pragma unroll
        for (int nt = 0; nt < 4; ++nt)
            acc[mt][nt] = v8f{0.f,0.f,0.f,0.f,0.f,0.f,0.f,0.f};

    for (int k0 = 0; k0 < DIN; k0 += 32) {
        __syncthreads();
        // stage A tile 128x32 (f32 -> bf16)
        #pragma unroll
        for (int it = 0; it < 4; ++it) {
            int idx = t + it * 256;            // 1024 float4s
            int r   = idx >> 3;                // 8 float4 per row
            int c4  = idx & 7;
            v4f f = *(const v4f*)(X + (size_t)(m0 + r) * DIN + k0 + c4 * 4);
            v4bf hb;
            #pragma unroll
            for (int j = 0; j < 4; ++j) hb[j] = f2bf(f[j]);
            *(v4bf*)(Alds + r * 32 + c4 * 4) = hb;
        }
        // stage B tile 32x128, transposed into [col][k]
        #pragma unroll
        for (int it = 0; it < 4; ++it) {
            int idx = t + it * 256;
            int kr  = idx >> 5;                // 32 float4 per k-row
            int c4  = idx & 31;
            v4f f = *(const v4f*)(W + (size_t)(k0 + kr) * DOUT + n0 + c4 * 4);
            #pragma unroll
            for (int j = 0; j < 4; ++j)
                Blds[(c4 * 4 + j) * 32 + kr] = f2bf(f[j]);
        }
        if (k0 + 32 < DIN) {   // global_prefetch_b8 for next K-step tiles
            __builtin_prefetch(X + (size_t)(m0 + (t >> 1)) * DIN + k0 + 32, 0, 1);
            __builtin_prefetch(W + (size_t)(k0 + 32 + (t >> 3)) * DOUT + n0, 0, 1);
        }
        __syncthreads();

        // fragments per ISA 16-bit A/B layouts, 8 WMMAs
        v16bf afr[2], bfr[4];
        #pragma unroll
        for (int mt = 0; mt < 2; ++mt) {
            const bf16* p = Alds + (wm * 32 + mt * 16 + lrow) * 32 + g * 8;
            afr[mt] = join8(*(const v8bf*)p, *(const v8bf*)(p + 16));
        }
        #pragma unroll
        for (int nt = 0; nt < 4; ++nt) {
            const bf16* p = Blds + (wn * 64 + nt * 16 + lrow) * 32 + g * 8;
            bfr[nt] = join8(*(const v8bf*)p, *(const v8bf*)(p + 16));
        }
        #pragma unroll
        for (int mt = 0; mt < 2; ++mt)
            #pragma unroll
            for (int nt = 0; nt < 4; ++nt)
                acc[mt][nt] = wmma_bf16(afr[mt], bfr[nt], acc[mt][nt]);
    }

    // epilogue
    #pragma unroll
    for (int mt = 0; mt < 2; ++mt) {
        #pragma unroll
        for (int nt = 0; nt < 4; ++nt) {
            if (TRANS) {
                // [B,H,D,S]: lane's 8 rows are consecutive s -> one 16B store
                int row0 = m0 + wm * 32 + mt * 16 + 8 * g;   // multiple of 8
                int col  = n0 + wn * 64 + nt * 16 + lrow;
                int b = row0 >> 11, s0 = row0 & 2047;
                int h = col >> 6,   d  = col & 63;
                v8bf pk;
                #pragma unroll
                for (int i = 0; i < 8; ++i) pk[i] = f2bf(acc[mt][nt][i]);
                *(v8bf*)(Y + (((size_t)(b * NHEAD + h) * HDIM + d) * SEQ + s0)) = pk;
            } else {
                #pragma unroll
                for (int i = 0; i < 8; ++i) {
                    int row = m0 + wm * 32 + mt * 16 + i + 8 * g;
                    int col = n0 + wn * 64 + nt * 16 + lrow;
                    int b = row >> 11, s = row & 2047;
                    int h = col >> 6,  d = col & 63;
                    Y[((size_t)(b * NHEAD + h) * SEQ + s) * HDIM + d] =
                        f2bf(acc[mt][nt][i]);
                }
            }
        }
    }
}

// ---------------------------------------------------------------------------
// Flash attention. Block = 256 threads (8 waves) covers 128 q-rows of one
// (b,h); each wave owns 16 q-rows. K/V streamed in 64-key chunks, double-
// buffered in LDS via the Tensor Data Mover (TENSORcnt-tracked).
// ---------------------------------------------------------------------------
__global__ __launch_bounds__(256)
void attn_fwd(const bf16* __restrict__ Qp, const bf16* __restrict__ Kp,
              const bf16* __restrict__ Vt, const float* __restrict__ vmask,
              const float* __restrict__ qmask, float* __restrict__ out)
{
    __shared__ __align__(16) bf16 Klds[2 * 64 * 64];  // [buf][kj][d]
    __shared__ __align__(16) bf16 Vlds[2 * 64 * 64];  // [buf][d][kj]
    __shared__ __align__(16) bf16 Plds[8 * 16 * 64];  // per-wave P tiles

    const int t    = threadIdx.x;
    const int lane = t & 31;
    const int wid  = t >> 5;
    const int lrow = lane & 15;
    const int g    = lane >> 4;
    const int bh   = blockIdx.y;
    const int b    = bh >> 4;
    const int h    = bh & 15;
    const int q0   = blockIdx.x * 128 + wid * 16;

    bf16* Pw = Plds + wid * 16 * 64;

    // Q fragments (held in registers for the whole kernel)
    const bf16* qbase = Qp + ((size_t)bh * SEQ + q0 + lrow) * HDIM;
    v16bf aq[2];
    #pragma unroll
    for (int ks = 0; ks < 2; ++ks) {
        const bf16* p = qbase + ks * 32 + g * 8;
        aq[ks] = join8(*(const v8bf*)p, *(const v8bf*)(p + 16));
    }

    float mrow[8], lsum[8];
    v8f acco[4];
    #pragma unroll
    for (int i = 0; i < 8; ++i) { mrow[i] = -3.0e38f; lsum[i] = 0.f; }
    #pragma unroll
    for (int dt = 0; dt < 4; ++dt)
        acco[dt] = v8f{0.f,0.f,0.f,0.f,0.f,0.f,0.f,0.f};

    const float scale = 0.125f;   // 1/sqrt(64)

#if USE_TDM
    // K chunk: 64 rows x 64 bf16, row stride 64 elems (fully contiguous 8KB)
    // V chunk: 64 rows x 64 bf16, row stride SEQ elems (strided 2D tile)
    const i32x8 g1k = tdm_group1_2d(64, 64, (unsigned long long)HDIM, 64, 64);
    const i32x8 g1v = tdm_group1_2d(64, 64, (unsigned long long)SEQ, 2048, 64);
    const unsigned long long kga =
        (unsigned long long)(uintptr_t)(Kp + (size_t)bh * SEQ * HDIM);
    const unsigned long long vga =
        (unsigned long long)(uintptr_t)(Vt + (size_t)bh * HDIM * SEQ);
    const unsigned ldsK = lds_offset_of(Klds);
    const unsigned ldsV = lds_offset_of(Vlds);

    if (wid == 0) {   // prologue: DMA chunk 0 into buffer 0
        tdm_load_2d(ldsK, kga, g1k);
        tdm_load_2d(ldsV, vga, g1v);
    }
#endif

    for (int kc = 0; kc < SEQ; kc += 64) {
        const int buf = (kc >> 6) & 1;
        const bf16* Kb = Klds + buf * 4096;
        const bf16* Vb = Vlds + buf * 4096;

#if USE_TDM
        if (wid == 0) __builtin_amdgcn_s_wait_tensorcnt(0);  // chunk kc landed
        __syncthreads();                                     // visible to all
        if (wid == 0 && kc + 64 < SEQ) {                     // DMA next chunk
            tdm_load_2d(ldsK + (buf ^ 1) * 8192,
                        kga + (size_t)(kc + 64) * HDIM * 2, g1k);
            tdm_load_2d(ldsV + (buf ^ 1) * 8192,
                        vga + (size_t)(kc + 64) * 2, g1v);
        }
#else
        __syncthreads();
        #pragma unroll
        for (int it = 0; it < 2; ++it) {
            int idx = t + it * 256;            // 512 v8bf per matrix
            int r   = idx >> 3;
            int c   = (idx & 7) * 8;
            *(v8bf*)(const_cast<bf16*>(Kb) + r * 64 + c) =
                *(const v8bf*)(Kp + ((size_t)bh * SEQ + kc + r) * HDIM + c);
            *(v8bf*)(const_cast<bf16*>(Vb) + r * 64 + c) =
                *(const v8bf*)(Vt + ((size_t)bh * HDIM + r) * SEQ + kc + c);
        }
        __syncthreads();
#endif

        // ---- scores: S = Q K^T * scale + mask-bias --------------------
        v8f sreg[4];
        #pragma unroll
        for (int nt = 0; nt < 4; ++nt) {
            v8f s = v8f{0.f,0.f,0.f,0.f,0.f,0.f,0.f,0.f};
            #pragma unroll
            for (int ks = 0; ks < 2; ++ks) {
                const bf16* p = Kb + (nt * 16 + lrow) * 64 + ks * 32 + g * 8;
                v16bf bk = join8(*(const v8bf*)p, *(const v8bf*)(p + 16));
                s = wmma_bf16(aq[ks], bk, s);
            }
            float vm   = vmask[(size_t)b * SEQ + kc + nt * 16 + lrow];
            float bias = (1.0f - vm) * -1.0e10f;
            #pragma unroll
            for (int i = 0; i < 8; ++i) s[i] = s[i] * scale + bias;
            sreg[nt] = s;
        }

        // ---- online softmax (each row lives on one half-wave) ---------
        #pragma unroll
        for (int i = 0; i < 8; ++i) {
            float cm = fmaxf(fmaxf(sreg[0][i], sreg[1][i]),
                             fmaxf(sreg[2][i], sreg[3][i]));
            cm = fmaxf(cm, __shfl_xor(cm, 1));
            cm = fmaxf(cm, __shfl_xor(cm, 2));
            cm = fmaxf(cm, __shfl_xor(cm, 4));
            cm = fmaxf(cm, __shfl_xor(cm, 8));
            float mnew  = fmaxf(mrow[i], cm);
            float alpha = __expf(mrow[i] - mnew);
            mrow[i] = mnew;
            float rs = 0.f;
            #pragma unroll
            for (int nt = 0; nt < 4; ++nt) {
                float p = __expf(sreg[nt][i] - mnew);
                rs += p;
                Pw[(i + 8 * g) * 64 + nt * 16 + lrow] = f2bf(p);  // C->A relayout
            }
            rs += __shfl_xor(rs, 1);
            rs += __shfl_xor(rs, 2);
            rs += __shfl_xor(rs, 4);
            rs += __shfl_xor(rs, 8);
            lsum[i] = lsum[i] * alpha + rs;
            #pragma unroll
            for (int dt = 0; dt < 4; ++dt) acco[dt][i] *= alpha;
        }
        __syncthreads();   // make per-wave P tile visible for ds reads

        // ---- O += P * V ----------------------------------------------
        #pragma unroll
        for (int ks = 0; ks < 2; ++ks) {
            const bf16* pp = Pw + lrow * 64 + ks * 32 + g * 8;
            v16bf ap = join8(*(const v8bf*)pp, *(const v8bf*)(pp + 16));
            #pragma unroll
            for (int dt = 0; dt < 4; ++dt) {
                const bf16* vp = Vb + (dt * 16 + lrow) * 64 + ks * 32 + g * 8;
                v16bf bv = join8(*(const v8bf*)vp, *(const v8bf*)(vp + 16));
                acco[dt] = wmma_bf16(ap, bv, acco[dt]);
            }
        }
    }

    // ---- epilogue: normalize, q_mask, scatter to [B,S,H*D] fp32 -------
    #pragma unroll
    for (int i = 0; i < 8; ++i) {
        int   row = q0 + i + 8 * g;
        float f   = (1.0f / lsum[i]) * qmask[(size_t)b * SEQ + row];
        #pragma unroll
        for (int dt = 0; dt < 4; ++dt) {
            out[((size_t)b * SEQ + row) * DOUT + h * HDIM + dt * 16 + lrow] =
                acco[dt][i] * f;
        }
    }
}

// ---------------------------------------------------------------------------
extern "C" void kernel_launch(void* const* d_in, const int* in_sizes, int n_in,
                              void* d_out, int out_size, void* d_ws, size_t ws_size,
                              hipStream_t stream)
{
    (void)in_sizes; (void)n_in; (void)out_size; (void)ws_size;

    const float* q      = (const float*)d_in[0];
    const float* k      = (const float*)d_in[1];
    const float* v      = (const float*)d_in[2];
    const float* v_mask = (const float*)d_in[3];
    const float* q_mask = (const float*)d_in[4];
    const float* Wq     = (const float*)d_in[5];
    const float* Wk     = (const float*)d_in[6];
    const float* Wv     = (const float*)d_in[7];
    float* out = (float*)d_out;

    const size_t nproj = (size_t)BATCH * NHEAD * SEQ * HDIM;   // 4M elems
    bf16* Qp = (bf16*)d_ws;
    bf16* Kp = Qp + nproj;
    bf16* Vt = Kp + nproj;

    dim3 gblk(DOUT / 128, MROWS / 128);          // (8, 32)
    proj_gemm<0><<<gblk, 256, 0, stream>>>(q, Wq, Qp);
    proj_gemm<0><<<gblk, 256, 0, stream>>>(k, Wk, Kp);
    proj_gemm<1><<<gblk, 256, 0, stream>>>(v, Wv, Vt);

    dim3 gattn(SEQ / 128, BATCH * NHEAD);        // (16, 32)
    attn_fwd<<<gattn, 256, 0, stream>>>(Qp, Kp, Vt, v_mask, q_mask, out);
}
